// DGCNN_74268574483105
// MI455X (gfx1250) — matrix operand order repaired and checked by reading, not compiled, not run on hardware
//
#include <hip/hip_runtime.h>
#include <hip/hip_bf16.h>

typedef __attribute__((ext_vector_type(16))) _Float16 v16h;
typedef __attribute__((ext_vector_type(8)))  _Float16 v8h_t;
typedef __attribute__((ext_vector_type(8)))  float    v8f;

#define BN_SCALE 0.9999950000374997f   /* 1/sqrt(1+1e-5) */
#define NPTS     16384                 /* 8 * 2048 */

__device__ __forceinline__ v8f wmma16(v16h a, v16h b, v8f c) {
    return __builtin_amdgcn_wmma_f32_16x16x32_f16(false, a, false, b,
                                                  (short)0, c, false, false);
}

// ---------------------------------------------------------------------------
// Transpose x [8,3,2048] -> ptsf f32 [16384,3] and ptsh f16 [16384,8] (padded)
// ---------------------------------------------------------------------------
__global__ void k_prep_pts(const float* __restrict__ x,
                           float* __restrict__ ptsf,
                           _Float16* __restrict__ ptsh) {
    int p = blockIdx.x * blockDim.x + threadIdx.x;
    if (p >= NPTS) return;
    int b = p >> 11, n = p & 2047;
    float v0 = x[(size_t)b * 3 * 2048 + 0 * 2048 + n];
    float v1 = x[(size_t)b * 3 * 2048 + 1 * 2048 + n];
    float v2 = x[(size_t)b * 3 * 2048 + 2 * 2048 + n];
    ptsf[p * 3 + 0] = v0; ptsf[p * 3 + 1] = v1; ptsf[p * 3 + 2] = v2;
    ptsh[p * 8 + 0] = (_Float16)v0;
    ptsh[p * 8 + 1] = (_Float16)v1;
    ptsh[p * 8 + 2] = (_Float16)v2;
    ptsh[p * 8 + 3] = (_Float16)0.f;
    ptsh[p * 8 + 4] = (_Float16)0.f;
    ptsh[p * 8 + 5] = (_Float16)0.f;
    ptsh[p * 8 + 6] = (_Float16)0.f;
    ptsh[p * 8 + 7] = (_Float16)0.f;
}

// ---------------------------------------------------------------------------
// Split edge-conv weight w [Cout, 2C] into wA = w[:, :C] and wD = w[:, C:] - wA
// ---------------------------------------------------------------------------
__global__ void k_prep_edge_w(const float* __restrict__ w, int C, int Cout,
                              _Float16* __restrict__ wA,
                              _Float16* __restrict__ wD) {
    int t = blockIdx.x * blockDim.x + threadIdx.x;
    if (t >= C * Cout) return;
    int o = t / C, c = t % C;
    float a  = w[(size_t)o * 2 * C + c];
    float bb = w[(size_t)o * 2 * C + C + c];
    wA[(size_t)o * C + c] = (_Float16)a;
    wD[(size_t)o * C + c] = (_Float16)(bb - a);
}

__global__ void k_f2h(const float* __restrict__ src,
                      _Float16* __restrict__ dst, int n) {
    int t = blockIdx.x * blockDim.x + threadIdx.x;
    if (t < n) dst[t] = (_Float16)src[t];
}

// ---------------------------------------------------------------------------
// KNN: one wave per point (4 points / 128-thread block, all in one batch).
// ---------------------------------------------------------------------------
__global__ void __launch_bounds__(128)
k_knn(const _Float16* __restrict__ feat, int ldf, int C,
      int* __restrict__ idxOut) {
    __shared__ float    d2s[4][2048];        // 32 KB
    __shared__ _Float16 tile[32 * 256];      // 16 KB
    __shared__ float    cfeat[4][256];       //  4 KB
    const int tid = threadIdx.x, w = tid >> 5, lane = tid & 31;
    const int p0 = blockIdx.x * 4;
    const int b  = p0 >> 11;

    for (int t = tid; t < 4 * C; t += 128) {
        int p = t / C, c = t % C;
        cfeat[p][c] = (float)feat[(size_t)(p0 + p) * ldf + c];
    }
    __syncthreads();

    const int chunks = C >> 3;
    for (int j0 = 0; j0 < 2048; j0 += 32) {
        for (int t = tid; t < 32 * chunks; t += 128) {
            int r = t / chunks, cc = (t % chunks) << 3;
            *(v8h_t*)&tile[r * C + cc] =
                *(const v8h_t*)&feat[(size_t)(b * 2048 + j0 + r) * ldf + cc];
        }
        __syncthreads();
        float acc = 0.f;
        for (int c = 0; c < C; ++c) {
            float d = cfeat[w][c] - (float)tile[lane * C + c];
            acc += d * d;
        }
        d2s[w][j0 + lane] = acc;
        __syncthreads();
    }

    for (int t = 0; t < 20; ++t) {
        float best = 3.3e38f; int bidx = 0x7fffffff;
        for (int j = lane; j < 2048; j += 32) {
            float v = d2s[w][j];
            if (v < best || (v == best && j < bidx)) { best = v; bidx = j; }
        }
        for (int off = 16; off > 0; off >>= 1) {
            float ov = __shfl_down(best, off, 32);
            int   oi = __shfl_down(bidx, off, 32);
            if (ov < best || (ov == best && oi < bidx)) { best = ov; bidx = oi; }
        }
        bidx = __shfl(bidx, 0, 32);
        if (lane == 0) {
            idxOut[(size_t)(p0 + w) * 20 + t] = b * 2048 + bidx;
            d2s[w][bidx] = 3.3e38f;
        }
    }
}

// ---------------------------------------------------------------------------
// EdgeConv1 (Cin=3): scalar, writes f16 into xcat columns [0,64)
// ---------------------------------------------------------------------------
__global__ void __launch_bounds__(64)
k_conv1(const float* __restrict__ ptsf, const int* __restrict__ idx,
        const float* __restrict__ w1, const float* __restrict__ g1,
        const float* __restrict__ b1, _Float16* __restrict__ xcat) {
    int p = blockIdx.x, o = threadIdx.x;
    float c0 = ptsf[p * 3 + 0], c1 = ptsf[p * 3 + 1], c2 = ptsf[p * 3 + 2];
    float wa0 = w1[o * 6 + 0], wa1 = w1[o * 6 + 1], wa2 = w1[o * 6 + 2];
    float wb0 = w1[o * 6 + 3], wb1 = w1[o * 6 + 4], wb2 = w1[o * 6 + 5];
    float ctr = (wb0 - wa0) * c0 + (wb1 - wa1) * c1 + (wb2 - wa2) * c2;
    float s = g1[o] * BN_SCALE, bb = b1[o];
    float best = 0.f;   // == max over relu values
    for (int k = 0; k < 20; ++k) {
        int j = idx[p * 20 + k];
        float z = wa0 * ptsf[j * 3 + 0] + wa1 * ptsf[j * 3 + 1] +
                  wa2 * ptsf[j * 3 + 2] + ctr;
        best = fmaxf(best, fmaf(z, s, bb));
    }
    xcat[(size_t)p * 512 + o] = (_Float16)best;
}

// ---------------------------------------------------------------------------
// WMMA NT-GEMM with 2x2 register blocking: each wave computes a 32x32 tile.
// block = 128 threads (4 waves stacked in M -> 128-row block tile);
// grid = (M/128, N/32).
// mode 0: store f32 to out (ldc)
// mode 2: v = relu(bn(acc)); atomic-max into hmax, atomic-add into hsum
// ---------------------------------------------------------------------------
__global__ void __launch_bounds__(128)
k_gemm_nt(const _Float16* __restrict__ A, int lda,
          const _Float16* __restrict__ W, int ldw, int Kdim,
          float* __restrict__ out, int ldc,
          const float* __restrict__ g, const float* __restrict__ bta,
          float* __restrict__ hmax, float* __restrict__ hsum, int mode) {
    const int w = threadIdx.x >> 5, lane = threadIdx.x & 31;
    const int m0 = (blockIdx.x * 4 + w) * 32;
    const int n0 = blockIdx.y * 32;
    const int hlf = (lane < 16) ? 0 : 1;

    v8f acc[2][2];
#pragma unroll
    for (int i = 0; i < 2; ++i)
#pragma unroll
        for (int j = 0; j < 2; ++j)
#pragma unroll
            for (int r = 0; r < 8; ++r) acc[i][j][r] = 0.f;

    const _Float16* arow0 = A + (size_t)(m0 + (lane & 15)) * lda;
    const _Float16* arow1 = arow0 + (size_t)16 * lda;
    const _Float16* brow0 = W + (size_t)(n0 + (lane & 15)) * ldw;
    const _Float16* brow1 = brow0 + (size_t)16 * ldw;
    const int aoff_lo = hlf ? 16 : 0, aoff_hi = hlf ? 24 : 8;
    const int boff    = hlf ? 16 : 0;

    for (int k0 = 0; k0 < Kdim; k0 += 32) {
        if (k0 + 256 < Kdim) {   // stream-ahead hint for the A operand
            __builtin_prefetch(arow0 + k0 + 256, 0, 0);
            __builtin_prefetch(arow1 + k0 + 256, 0, 0);
        }
        v8h_t a0lo = *(const v8h_t*)(arow0 + k0 + aoff_lo);
        v8h_t a0hi = *(const v8h_t*)(arow0 + k0 + aoff_hi);
        v8h_t a1lo = *(const v8h_t*)(arow1 + k0 + aoff_lo);
        v8h_t a1hi = *(const v8h_t*)(arow1 + k0 + aoff_hi);
        v16h af0, af1;
#pragma unroll
        for (int i = 0; i < 8; ++i) {
            af0[i] = a0lo[i]; af0[i + 8] = a0hi[i];
            af1[i] = a1lo[i]; af1[i + 8] = a1hi[i];
        }
        v16h bf0 = *(const v16h*)(brow0 + k0 + boff);
        v16h bf1 = *(const v16h*)(brow1 + k0 + boff);
        acc[0][0] = wmma16(af0, bf0, acc[0][0]);
        acc[0][1] = wmma16(af0, bf1, acc[0][1]);
        acc[1][0] = wmma16(af1, bf0, acc[1][0]);
        acc[1][1] = wmma16(af1, bf1, acc[1][1]);
    }

    if (mode == 0) {
#pragma unroll
        for (int mt = 0; mt < 2; ++mt)
#pragma unroll
            for (int nt = 0; nt < 2; ++nt) {
                const int col = n0 + nt * 16 + (lane & 15);
#pragma unroll
                for (int r = 0; r < 8; ++r)
                    out[(size_t)(m0 + mt * 16 + r + hlf * 8) * ldc + col] =
                        acc[mt][nt][r];
            }
    } else {
        const int batch = m0 >> 11;   // 32-row tile stays within one batch
#pragma unroll
        for (int nt = 0; nt < 2; ++nt) {
            const int col = n0 + nt * 16 + (lane & 15);
            float s = g[col] * BN_SCALE, bb = bta[col];
            float vmax = 0.f, vsum = 0.f;
#pragma unroll
            for (int mt = 0; mt < 2; ++mt)
#pragma unroll
                for (int r = 0; r < 8; ++r) {
                    float v = fmaxf(fmaf(acc[mt][nt][r], s, bb), 0.f);
                    vmax = fmaxf(vmax, v);
                    vsum += v;
                }
            float om = __shfl_down(vmax, 16, 32);
            float os = __shfl_down(vsum, 16, 32);
            if (lane < 16) {
                vmax = fmaxf(vmax, om);
                vsum += os;
                atomicMax((unsigned int*)&hmax[batch * 1024 + col],
                          __float_as_uint(vmax));     // all values >= 0
                atomicAdd(&hsum[batch * 1024 + col], vsum);
            }
        }
    }
}

// ---------------------------------------------------------------------------
// Fused neighbor EdgeConv (WMMA): 4 points/block -> 80 gathered rows =
// exactly 5 M-tiles of 16. Gather neighbor features to LDS once; per 16-col
// N-tile run K-loop of WMMAs, stage 80x16 result to LDS, reduce max over the
// 20 neighbors with center term + BN + ReLU, write f16 to xcat out-slice.
// ---------------------------------------------------------------------------
__global__ void __launch_bounds__(128)
k_edge_fused(const _Float16* __restrict__ xin, int colIn, int C,
             const int* __restrict__ idx, const float* __restrict__ Gc,
             int Cout, const _Float16* __restrict__ wA,
             const float* __restrict__ g, const float* __restrict__ bta,
             _Float16* __restrict__ xout, int colOut) {
    __shared__ _Float16 As[80 * 256];        // 40 KB max
    __shared__ float    Rs[4][80 * 16];      // 20 KB
    const int tid = threadIdx.x, w = tid >> 5, lane = tid & 31;
    const int p0 = blockIdx.x * 4;
    const int chunks = C >> 3;

    for (int t = tid; t < 80 * chunks; t += 128) {
        int r = t / chunks, cc = (t % chunks) << 3;
        int src = idx[(size_t)(p0 + r / 20) * 20 + (r % 20)];
        *(v8h_t*)&As[r * C + cc] =
            *(const v8h_t*)&xin[(size_t)src * 512 + colIn + cc];
    }
    __syncthreads();

    const int hlf = (lane < 16) ? 0 : 1;
    const int ntiles = Cout >> 4;
    for (int nt = w; nt < ntiles; nt += 4) {
        const int n0 = nt << 4;
        v8f acc[5];
#pragma unroll
        for (int mt = 0; mt < 5; ++mt)
#pragma unroll
            for (int i = 0; i < 8; ++i) acc[mt][i] = 0.f;

        const _Float16* brow = wA + (size_t)(n0 + (lane & 15)) * C;
        for (int k0 = 0; k0 < C; k0 += 32) {
            v16h bf = *(const v16h*)(brow + k0 + (hlf ? 16 : 0));
#pragma unroll
            for (int mt = 0; mt < 5; ++mt) {
                const _Float16* ap = &As[(mt * 16 + (lane & 15)) * C + k0];
                v8h_t alo = *(const v8h_t*)(ap + (hlf ? 16 : 0));
                v8h_t ahi = *(const v8h_t*)(ap + (hlf ? 24 : 8));
                v16h af;
#pragma unroll
                for (int i = 0; i < 8; ++i) { af[i] = alo[i]; af[i + 8] = ahi[i]; }
                acc[mt] = wmma16(af, bf, acc[mt]);
            }
        }
#pragma unroll
        for (int mt = 0; mt < 5; ++mt)
#pragma unroll
            for (int r = 0; r < 8; ++r)
                Rs[w][(mt * 16 + r + hlf * 8) * 16 + (lane & 15)] = acc[mt][r];

        // per-wave LDS RAW within the same wave: in-order DS, no barrier needed
#pragma unroll
        for (int q = 0; q < 2; ++q) {
            int f = q * 32 + lane;
            int p = f >> 4, c = f & 15;
            float gcv = Gc[(size_t)(p0 + p) * Cout + n0 + c];
            float s = g[n0 + c] * BN_SCALE, bb = bta[n0 + c];
            float best = 0.f;   // relu floor => max over relu values
            for (int r = 0; r < 20; ++r) {
                float v = fmaf(Rs[w][(p * 20 + r) * 16 + c] + gcv, s, bb);
                best = fmaxf(best, v);
            }
            xout[(size_t)(p0 + p) * 512 + colOut + n0 + c] = (_Float16)best;
        }
    }
}

// ---------------------------------------------------------------------------
// FC head (M=8, scalar)
// ---------------------------------------------------------------------------
__global__ void k_fc1(const float* __restrict__ hmax, const float* __restrict__ hsum,
                      const float* __restrict__ wl1, const float* __restrict__ g6,
                      const float* __restrict__ b6, float* __restrict__ h1) {
    int b = blockIdx.x, o = threadIdx.x;          // 512 threads
    const float inv = 1.0f / 2048.0f;
    float acc = 0.f;
    for (int c = 0; c < 1024; ++c)
        acc = fmaf(hmax[b * 1024 + c], wl1[(size_t)o * 2048 + c], acc);
    for (int c = 0; c < 1024; ++c)
        acc = fmaf(hsum[b * 1024 + c] * inv, wl1[(size_t)o * 2048 + 1024 + c], acc);
    float v = fmaf(acc, g6[o] * BN_SCALE, b6[o]);
    h1[b * 512 + o] = (v > 0.f) ? v : 0.2f * v;
}

__global__ void k_fc2(const float* __restrict__ h1, const float* __restrict__ wl2,
                      const float* __restrict__ bl2, const float* __restrict__ g7,
                      const float* __restrict__ b7, float* __restrict__ h2) {
    int b = blockIdx.x, o = threadIdx.x;          // 256 threads
    float acc = bl2[o];
    for (int c = 0; c < 512; ++c)
        acc = fmaf(h1[b * 512 + c], wl2[(size_t)o * 512 + c], acc);
    float v = fmaf(acc, g7[o] * BN_SCALE, b7[o]);
    h2[b * 256 + o] = (v > 0.f) ? v : 0.2f * v;
}

__global__ void k_fc3(const float* __restrict__ h2, const float* __restrict__ wl3,
                      const float* __restrict__ bl3, float* __restrict__ out) {
    int b = blockIdx.x, o = threadIdx.x;
    if (o >= 40) return;
    float acc = bl3[o];
    for (int c = 0; c < 256; ++c)
        acc = fmaf(h2[b * 256 + c], wl3[(size_t)o * 256 + c], acc);
    out[b * 40 + o] = acc;
}

// ---------------------------------------------------------------------------
extern "C" void kernel_launch(void* const* d_in, const int* in_sizes, int n_in,
                              void* d_out, int out_size, void* d_ws, size_t ws_size,
                              hipStream_t stream) {
    const float* x   = (const float*)d_in[0];
    const float* w1  = (const float*)d_in[1];
    const float* g1  = (const float*)d_in[2];
    const float* b1  = (const float*)d_in[3];
    const float* w2  = (const float*)d_in[4];
    const float* g2  = (const float*)d_in[5];
    const float* b2  = (const float*)d_in[6];
    const float* w3  = (const float*)d_in[7];
    const float* g3  = (const float*)d_in[8];
    const float* b3  = (const float*)d_in[9];
    const float* w4  = (const float*)d_in[10];
    const float* g4  = (const float*)d_in[11];
    const float* b4  = (const float*)d_in[12];
    const float* w5  = (const float*)d_in[13];
    const float* g5  = (const float*)d_in[14];
    const float* b5  = (const float*)d_in[15];
    const float* wl1 = (const float*)d_in[16];
    const float* g6  = (const float*)d_in[17];
    const float* b6  = (const float*)d_in[18];
    const float* wl2 = (const float*)d_in[19];
    const float* bl2 = (const float*)d_in[20];
    const float* g7  = (const float*)d_in[21];
    const float* b7  = (const float*)d_in[22];
    const float* wl3 = (const float*)d_in[23];
    const float* bl3 = (const float*)d_in[24];

    char* ws = (char*)d_ws;
    size_t off = 0;
    auto alloc = [&](size_t bytes) -> void* {
        void* p = ws + off;
        off = (off + bytes + 255) & ~(size_t)255;
        return p;
    };
    float*    ptsf = (float*)alloc((size_t)NPTS * 3 * 4);
    _Float16* ptsh = (_Float16*)alloc((size_t)NPTS * 8 * 2);
    _Float16* xcat = (_Float16*)alloc((size_t)NPTS * 512 * 2);
    int*      idx  = (int*)alloc((size_t)NPTS * 20 * 4);
    float*    Gc   = (float*)alloc((size_t)NPTS * 256 * 4);
    float*    hmax = (float*)alloc(8 * 1024 * 4);
    float*    hsum = (float*)alloc(8 * 1024 * 4);
    float*    h1b  = (float*)alloc(8 * 512 * 4);
    float*    h2b  = (float*)alloc(8 * 256 * 4);
    _Float16* wA2  = (_Float16*)alloc(64 * 64 * 2);
    _Float16* wD2  = (_Float16*)alloc(64 * 64 * 2);
    _Float16* wA3  = (_Float16*)alloc(128 * 64 * 2);
    _Float16* wD3  = (_Float16*)alloc(128 * 64 * 2);
    _Float16* wA4  = (_Float16*)alloc(256 * 128 * 2);
    _Float16* wD4  = (_Float16*)alloc(256 * 128 * 2);
    _Float16* w5h  = (_Float16*)alloc((size_t)1024 * 512 * 2);

    hipMemsetAsync(hmax, 0, 8 * 1024 * 4, stream);
    hipMemsetAsync(hsum, 0, 8 * 1024 * 4, stream);

    k_prep_pts<<<NPTS / 256, 256, 0, stream>>>(x, ptsf, ptsh);
    k_prep_edge_w<<<(64 * 64 + 255) / 256, 256, 0, stream>>>(w2, 64, 64, wA2, wD2);
    k_prep_edge_w<<<(128 * 64 + 255) / 256, 256, 0, stream>>>(w3, 64, 128, wA3, wD3);
    k_prep_edge_w<<<(256 * 128 + 255) / 256, 256, 0, stream>>>(w4, 128, 256, wA4, wD4);
    k_f2h<<<(1024 * 512 + 255) / 256, 256, 0, stream>>>(w5, w5h, 1024 * 512);

    // EdgeConv 1 (scalar, Cin=3 -> 64)
    k_knn<<<NPTS / 4, 128, 0, stream>>>(ptsh, 8, 8, idx);
    k_conv1<<<NPTS, 64, 0, stream>>>(ptsf, idx, w1, g1, b1, xcat);

    // EdgeConv 2 (64 -> 64), output cols [64,128)
    k_knn<<<NPTS / 4, 128, 0, stream>>>(xcat + 0, 512, 64, idx);
    k_gemm_nt<<<dim3(NPTS / 128, 2), 128, 0, stream>>>(xcat + 0, 512, wD2, 64, 64,
        Gc, 64, nullptr, nullptr, nullptr, nullptr, 0);
    k_edge_fused<<<NPTS / 4, 128, 0, stream>>>(xcat, 0, 64, idx, Gc, 64,
        wA2, g2, b2, xcat, 64);

    // EdgeConv 3 (64 -> 128), output cols [128,256)
    k_knn<<<NPTS / 4, 128, 0, stream>>>(xcat + 64, 512, 64, idx);
    k_gemm_nt<<<dim3(NPTS / 128, 4), 128, 0, stream>>>(xcat + 64, 512, wD3, 64, 64,
        Gc, 128, nullptr, nullptr, nullptr, nullptr, 0);
    k_edge_fused<<<NPTS / 4, 128, 0, stream>>>(xcat, 64, 64, idx, Gc, 128,
        wA3, g3, b3, xcat, 128);

    // EdgeConv 4 (128 -> 256), output cols [256,512)
    k_knn<<<NPTS / 4, 128, 0, stream>>>(xcat + 128, 512, 128, idx);
    k_gemm_nt<<<dim3(NPTS / 128, 8), 128, 0, stream>>>(xcat + 128, 512, wD4, 128, 128,
        Gc, 256, nullptr, nullptr, nullptr, nullptr, 0);
    k_edge_fused<<<NPTS / 4, 128, 0, stream>>>(xcat, 128, 128, idx, Gc, 256,
        wA4, g4, b4, xcat, 256);

    // conv5 (512 -> 1024) + fused BN/ReLU + max/mean pooling via atomics
    k_gemm_nt<<<dim3(NPTS / 128, 32), 128, 0, stream>>>(xcat, 512, w5h, 512, 512,
        nullptr, 0, g5, b5, hmax, hsum, 2);

    // FC head
    k_fc1<<<8, 512, 0, stream>>>(hmax, hsum, wl1, g6, b6, h1b);
    k_fc2<<<8, 256, 0, stream>>>(h1b, wl2, bl2, g7, b7, h2b);
    k_fc3<<<8, 64, 0, stream>>>(h2b, wl3, bl3, (float*)d_out);
}
